// Attention_7206955123485
// MI455X (gfx1250) — matrix operand order repaired and checked by reading, not compiled, not run on hardware
//
#include <hip/hip_runtime.h>
#include <hip/hip_bf16.h>

// ---------------------------------------------------------------------------
// Flash-attention for MI455X (gfx1250, wave32, WMMA bf16).
// O = softmax(Q K^T, axis=1) * d^-0.5 @ V      (scale applied AFTER softmax,
// faithful to the reference; folded into the final normalization).
//
// Compute-bound (≈1000 FLOP/byte): 2 bf16 WMMAs per 16x16 tile step, f32
// accumulation, online softmax. Q staged once in LDS in A-operand order;
// V staged per key-block in B-operand order (transpose done during staging);
// K B-operands gathered straight from the L2-resident K matrix; P exchanged
// between the 4 column-waves of a row-group through LDS in A-operand order.
// ---------------------------------------------------------------------------

typedef __attribute__((ext_vector_type(16))) __bf16 v16bf;
typedef __attribute__((ext_vector_type(8)))  float  v8f;
typedef __attribute__((ext_vector_type(4)))  float  vf4;
typedef __attribute__((ext_vector_type(2)))  float  vf2;

#define N_Q   4096
#define N_K   4096
#define DIM   1024
#define VDIM  1024
#define BM    32          // query rows per workgroup (2 row-groups of 16)
#define BK    64          // keys per iteration (4 S-tiles, one per col-wave)
#define NTHREADS 256      // 8 waves

// LDS layout (bytes)
#define QS_OFF  0                       // Q in A-operand order: 2 rg * 32 steps * 1KB
#define QS_SIZE (2*32*1024)
#define VS_OFF  (QS_OFF + QS_SIZE)      // V in B-operand order: 64 coltiles * 2 halves * 1KB
#define VS_SIZE (64*2*1024)
#define PB_OFF  (VS_OFF + VS_SIZE)      // P in A-operand order: 2 rg * 2 halves * 1KB
#define PB_SIZE (2*2*1024)
#define ST_OFF  (PB_OFF + PB_SIZE)      // row stats: [rg][cq][16] f32
#define ST_SIZE (2*4*16*4)
#define LDS_BYTES (ST_OFF + ST_SIZE)    // = 201216 B  (< 320 KB per WGP)

static __device__ inline unsigned pk_bf16(float a, float b) {
#if __has_builtin(__builtin_amdgcn_cvt_pk_bf16_f32)
  typedef __attribute__((ext_vector_type(2))) __bf16 v2bf;
  v2bf p = __builtin_amdgcn_cvt_pk_bf16_f32(a, b);
  return __builtin_bit_cast(unsigned, p);
#else
  __bf16 x = (__bf16)a, y = (__bf16)b;
  unsigned short ux = __builtin_bit_cast(unsigned short, x);
  unsigned short uy = __builtin_bit_cast(unsigned short, y);
  return (unsigned)ux | ((unsigned)uy << 16);
#endif
}

// Map an even K-index r (0..31) of a 16-bit A operand (16x32) to
// (lane-half, VGPR j) per the CDNA5 ISA A-layout table.
static __device__ inline void a_map(int r, int& half, int& j) {
  if      (r < 8)  { half = 0; j = r >> 1; }
  else if (r < 16) { half = 1; j = (r - 8) >> 1; }
  else if (r < 24) { half = 0; j = 4 + ((r - 16) >> 1); }
  else             { half = 1; j = 4 + ((r - 24) >> 1); }
}

extern "C" __global__ __launch_bounds__(NTHREADS)
void fa_cdna5_kernel(const float* __restrict__ Q, const float* __restrict__ K,
                     const float* __restrict__ V, float* __restrict__ O) {
  extern __shared__ char lds[];
  char*  Qs = lds + QS_OFF;
  char*  Vs = lds + VS_OFF;
  char*  Pb = lds + PB_OFF;
  float* St = (float*)(lds + ST_OFF);

  const int tid  = threadIdx.x;
  const int wave = tid >> 5;
  const int lane = tid & 31;
  const int rg   = wave >> 2;      // row-group 0..1 (16 query rows each)
  const int cq   = wave & 3;       // column wave 0..3 (256 output cols each)
  const int lh   = lane >> 4;      // lane half
  const int ln   = lane & 15;
  const int qb   = blockIdx.x * BM;

  // ---- Stage Q strip (32 rows x 1024) once into A-operand order ----------
  // 16384 bf16-pairs, 64 per thread; global reads are contiguous float2.
  for (int i = 0; i < 64; ++i) {
    int p    = tid + i * NTHREADS;
    int row  = p >> 9;             // 0..31
    int kd   = (p & 511) * 2;      // even d index
    int s    = kd >> 5;
    int half, j; a_map(kd & 31, half, j);
    vf2 qv   = *(const vf2*)(Q + (size_t)(qb + row) * DIM + kd);
    int rgq  = row >> 4, rloc = row & 15;
    *(unsigned*)(Qs + ((rgq * 32 + s) * 1024) + (rloc + 16 * half) * 32 + j * 4)
        = pk_bf16(qv[0], qv[1]);
  }
  __syncthreads();

  // ---- per-wave state ------------------------------------------------------
  v8f acc[16];                     // 16 col-tiles of 16x16 f32
  #pragma unroll
  for (int t = 0; t < 16; ++t)
    #pragma unroll
    for (int v = 0; v < 8; ++v) acc[t][v] = 0.0f;

  float m_prev[8], lsum[8];
  #pragma unroll
  for (int v = 0; v < 8; ++v) { m_prev[v] = -1.0e30f; lsum[v] = 0.0f; }

  // K gather base: this wave's 16 keys, its lane-half's two 8-float d chunks
  // (d = 8h..8h+7 and 16+8h..16+8h+7, A-table order; K is L2-resident, so the row stride is
  // served out of the 192MB L2.)
  const float* Krow0 = K + (size_t)(cq * 16 + ln) * DIM + lh * 8;

  // P-store swizzle for this wave (lane-uniform): kk32 = (cq&1)*16 + ln
  int pAh, pj; a_map((((cq & 1) << 4) | ln) & ~1, pAh, pj);

  for (int kb = 0; kb < N_K; kb += BK) {
    // ---- prefetch next key-block into the cache hierarchy -----------------
    if (kb + BK < N_K) {
      // next K rows for this wave's gather pattern (one line per lane)
      __builtin_prefetch(Krow0 + (size_t)(kb + BK) * DIM, 0, 1);
      // next V block rows, strided across the whole workgroup
      __builtin_prefetch(V + (size_t)(kb + BK + (tid >> 2)) * VDIM + (tid & 3) * 256, 0, 1);
    }

    // ---- Stage V block (64 keys x 1024 cols) into B-operand order ---------
    // thread reads float4 of rows k,k+1 (coalesced), pair-packs along k.
    for (int i = 0; i < 32; ++i) {
      int qp = tid + i * NTHREADS;         // 0..8191
      int k  = (qp >> 8) * 2;              // even key in 0..62
      int c4 = (qp & 255) * 4;             // col base
      vf4 v0 = *(const vf4*)(V + (size_t)(kb + k)     * VDIM + c4);
      vf4 v1 = *(const vf4*)(V + (size_t)(kb + k + 1) * VDIM + c4);
      int h  = k >> 5;
      int kk = k & 31;
      int bh, j; a_map(kk, bh, j);
      char* base = Vs + (((c4 >> 4) * 2 + h) * 1024) + j * 4;
      #pragma unroll
      for (int e = 0; e < 4; ++e)
        *(unsigned*)(base + (((c4 + e) & 15) + 16 * bh) * 32) = pk_bf16(v0[e], v1[e]);
    }
    __syncthreads();

    // ---- S = Q K^T : this wave's 16x16 tile over full d --------------------
    v8f s;
    #pragma unroll
    for (int v = 0; v < 8; ++v) s[v] = 0.0f;

    const float* Kp = Krow0 + (size_t)kb * DIM;
    #pragma unroll 4
    for (int si = 0; si < 32; ++si) {
      v16bf a = *(const v16bf*)(Qs + (rg * 32 + si) * 1024 + lane * 32);
      const vf4* kp = (const vf4*)(Kp + si * 32);
      vf4 f0 = kp[0], f1 = kp[1], f2 = kp[4], f3 = kp[5];
      v16bf b;
      #pragma unroll
      for (int e = 0; e < 4; ++e) {
        b[e]      = (__bf16)f0[e];
        b[4 + e]  = (__bf16)f1[e];
        b[8 + e]  = (__bf16)f2[e];
        b[12 + e] = (__bf16)f3[e];
      }
      s = __builtin_amdgcn_wmma_f32_16x16x32_bf16(false, a, false, b,
                                                  (short)0, s, false, false);
    }

    // ---- block row-max (reduce across 16 lanes of each half) --------------
    float bmax[8];
    #pragma unroll
    for (int v = 0; v < 8; ++v) {
      float x = s[v];
      x = fmaxf(x, __shfl_xor(x, 1, 32));
      x = fmaxf(x, __shfl_xor(x, 2, 32));
      x = fmaxf(x, __shfl_xor(x, 4, 32));
      x = fmaxf(x, __shfl_xor(x, 8, 32));
      bmax[v] = x;
    }
    if (ln == 0) {
      #pragma unroll
      for (int v = 0; v < 8; ++v)
        St[(rg * 4 + cq) * 16 + (v + 8 * lh)] = bmax[v];
    }
    __syncthreads();

    // ---- combine max across the 4 col-waves, rescale, exponentiate --------
    float mnew[8], alpha[8];
    #pragma unroll
    for (int v = 0; v < 8; ++v) {
      int row = v + 8 * lh;
      float m = m_prev[v];
      m = fmaxf(m, St[(rg * 4 + 0) * 16 + row]);
      m = fmaxf(m, St[(rg * 4 + 1) * 16 + row]);
      m = fmaxf(m, St[(rg * 4 + 2) * 16 + row]);
      m = fmaxf(m, St[(rg * 4 + 3) * 16 + row]);
      mnew[v]  = m;
      alpha[v] = __expf(m_prev[v] - m);
      m_prev[v] = m;
    }
    #pragma unroll
    for (int t = 0; t < 16; ++t)
      #pragma unroll
      for (int v = 0; v < 8; ++v) acc[t][v] *= alpha[v];

    #pragma unroll
    for (int v = 0; v < 8; ++v) {
      float p = __expf(s[v] - mnew[v]);
      s[v] = p;
      float x = p;
      x += __shfl_xor(x, 1, 32);
      x += __shfl_xor(x, 2, 32);
      x += __shfl_xor(x, 4, 32);
      x += __shfl_xor(x, 8, 32);
      lsum[v] = lsum[v] * alpha[v] + x;   // partial: this wave's 16 keys only
    }

    // ---- write P tile (bf16) into A-operand layout in LDS -----------------
    {
      char* pb = Pb + rg * 2048 + (cq >> 1) * 1024 + pj * 4 + (ln & 1) * 2;
      #pragma unroll
      for (int v = 0; v < 8; ++v) {
        int m = v + 8 * lh;
        *(__bf16*)(pb + (m + 16 * pAh) * 32) = (__bf16)s[v];
      }
    }
    __syncthreads();

    // ---- O += P @ V over this wave's 16 column tiles -----------------------
    #pragma unroll
    for (int h = 0; h < 2; ++h) {
      v16bf a = *(const v16bf*)(Pb + rg * 2048 + h * 1024 + lane * 32);
      #pragma unroll
      for (int t = 0; t < 16; ++t) {
        v16bf b = *(const v16bf*)(Vs + (((cq * 16 + t) * 2 + h) * 1024) + lane * 32);
        acc[t] = __builtin_amdgcn_wmma_f32_16x16x32_bf16(false, a, false, b,
                                                         (short)0, acc[t], false, false);
      }
    }
    __syncthreads();   // protect Vs/Pb/St before next iteration's staging
  }

  // ---- epilogue: combine partial row-sums across the 4 col-waves ----------
  if (ln == 0) {
    #pragma unroll
    for (int v = 0; v < 8; ++v)
      St[(rg * 4 + cq) * 16 + (v + 8 * lh)] = lsum[v];
  }
  __syncthreads();

  const float scale = 0.03125f;   // 1024^-0.5, applied after softmax (reference)
  float inv[8];
  #pragma unroll
  for (int v = 0; v < 8; ++v) {
    int row = v + 8 * lh;
    float tot = St[(rg * 4 + 0) * 16 + row] + St[(rg * 4 + 1) * 16 + row] +
                St[(rg * 4 + 2) * 16 + row] + St[(rg * 4 + 3) * 16 + row];
    inv[v] = scale / tot;
  }

  const int c0 = cq * 256;
  #pragma unroll
  for (int t = 0; t < 16; ++t) {
    int col = c0 + t * 16 + ln;
    #pragma unroll
    for (int v = 0; v < 8; ++v) {
      int row = qb + rg * 16 + v + 8 * lh;
      O[(size_t)row * VDIM + col] = acc[t][v] * inv[v];
    }
  }
}

extern "C" void kernel_launch(void* const* d_in, const int* in_sizes, int n_in,
                              void* d_out, int out_size, void* d_ws, size_t ws_size,
                              hipStream_t stream) {
  (void)in_sizes; (void)n_in; (void)out_size; (void)d_ws; (void)ws_size;
  const float* Q = (const float*)d_in[0];
  const float* K = (const float*)d_in[1];
  const float* V = (const float*)d_in[2];
  float* O = (float*)d_out;
  dim3 grid(N_Q / BM);      // 128 workgroups
  dim3 block(NTHREADS);     // 8 waves (wave32)
  fa_cdna5_kernel<<<grid, block, LDS_BYTES, stream>>>(Q, K, V, O);
}
